// WindowedAttentionBlock_38268158607447
// MI455X (gfx1250) — matrix operand order repaired and checked by reading, not compiled, not run on hardware
//
#include <hip/hip_runtime.h>
#include <hip/hip_bf16.h>

typedef __bf16 bf16_t;
typedef __attribute__((ext_vector_type(16))) __bf16 bf16x16;
typedef __attribute__((ext_vector_type(8)))  __bf16 bf16x8;
typedef __attribute__((ext_vector_type(8)))  float  floatx8;

#define NB      4
#define IMG_H   64
#define IMG_W   64
#define CH      1280
#define NHEADS  16
#define HD      80
#define WS_     14
#define NTOK    196
#define NWIN    100
#define NROWS   19600

// attention LDS geometry
#define A_NPAD  208   // 13 * 16 token tiles
#define A_QSTR  104   // q/k row stride (d padded to 96, +8 anti-bank-conflict)
#define A_VSTR  232   // vT row stride (token padded to 224, +8)
#define A_KPAD  224   // token dim padded for P@V (7 * 32)
#define A_SSTR  212   // S row stride (f32)
#define ATTN_SMEM (2*A_NPAD*A_QSTR*2 + 96*A_VSTR*2 + A_NPAD*A_SSTR*4 + 2*NTOK*WS_*2)

// window token row r -> flat image base offset (pixel * CH), or -1 if in padding
__device__ __forceinline__ int img_base(int r) {
    int w = r / NTOK, t = r - w * NTOK;
    int y = t / WS_,  x = t - y * WS_;
    int b = w / 25,   wi = w - b * 25;
    int Y = (wi / 5) * WS_ + y;
    int X = (wi % 5) * WS_ + x;
    if (Y >= IMG_H || X >= IMG_W) return -1;
    return ((b * IMG_H + Y) * IMG_W + X) * CH;
}

// ---- WMMA fragment helpers (LDS row base must be 16B aligned) ----
// A-matrix 16x32 bf16: lane row = lane&15; VGPR0..3 = K[koff..koff+7], VGPR4..7 = K[koff+16..koff+23], koff = half*8
__device__ __forceinline__ bf16x16 frag_a(const bf16_t* rowp, int k) {
    bf16x8 lo = *(const bf16x8*)(rowp + k);
    bf16x8 hi = *(const bf16x8*)(rowp + k + 16);
    bf16x16 r;
#pragma unroll
    for (int i = 0; i < 8; ++i) { r[i] = lo[i]; r[i + 8] = hi[i]; }
    return r;
}
// B-matrix 32x16 bf16: lane col = lane&15; VGPR0..3 = K[koff..koff+7], VGPR4..7 = K[koff+16..koff+23], koff = half*8
__device__ __forceinline__ bf16x16 frag_b(const bf16_t* rowp, int k) {
    bf16x8 lo = *(const bf16x8*)(rowp + k);
    bf16x8 hi = *(const bf16x8*)(rowp + k + 16);
    bf16x16 r;
#pragma unroll
    for (int i = 0; i < 8; ++i) { r[i] = lo[i]; r[i + 8] = hi[i]; }
    return r;
}

__device__ __forceinline__ floatx8 wmma_bf16(bf16x16 a, bf16x16 b, floatx8 c) {
    return __builtin_amdgcn_wmma_f32_16x16x32_bf16(false, a, false, b, (short)0, c, false, false);
}

// CDNA5 async copy: global -> LDS, 16B per lane, tracked by ASYNCcnt (no VGPR round-trip)
__device__ __forceinline__ void async_b128(unsigned lds_off, unsigned gvoff, const void* sbase) {
    asm volatile("global_load_async_to_lds_b128 %0, %1, %2"
                 :: "v"(lds_off), "v"(gvoff), "s"(sbase) : "memory");
}
__device__ __forceinline__ void wait_async0() {
    asm volatile("s_wait_asynccnt 0" ::: "memory");
}

// ---- weight convert + transpose: w[K][N] f32 -> wt[N][K] bf16 ----
__global__ __launch_bounds__(256) void wconv_kernel(const float* __restrict__ w,
                                                    bf16_t* __restrict__ wt, int K, int N) {
    size_t i = (size_t)blockIdx.x * 256 + threadIdx.x;
    if (i >= (size_t)K * N) return;
    int k = (int)(i / N);
    int n = (int)(i - (size_t)k * N);
    wt[(size_t)n * K + k] = (bf16_t)w[i];
}

// ---- LayerNorm over C=1280; src dense rows, or gather from image (imgp != null) ----
__global__ __launch_bounds__(256) void ln_kernel(const float* __restrict__ src,
                                                 const float* __restrict__ imgp,
                                                 const float* __restrict__ g,
                                                 const float* __restrict__ b,
                                                 bf16_t* __restrict__ out) {
    int r = blockIdx.x, tid = threadIdx.x;
    float vals[5];
    float s = 0.f, s2 = 0.f;
    int ob = imgp ? img_base(r) : 0;
#pragma unroll
    for (int i = 0; i < 5; ++i) {
        int c = tid + i * 256;
        float v;
        if (imgp) {
            v = (ob >= 0) ? imgp[ob + c] : 0.f;
        } else {
            v = src[(size_t)r * CH + c];
        }
        vals[i] = v; s += v; s2 += v * v;
    }
#pragma unroll
    for (int o = 16; o > 0; o >>= 1) { s += __shfl_xor(s, o, 32); s2 += __shfl_xor(s2, o, 32); }
    __shared__ float red[16];
    int wave = tid >> 5, lane = tid & 31;
    if (lane == 0) { red[wave] = s; red[8 + wave] = s2; }
    __syncthreads();
    float ts = 0.f, ts2 = 0.f;
#pragma unroll
    for (int i = 0; i < 8; ++i) { ts += red[i]; ts2 += red[8 + i]; }
    float mean = ts * (1.f / CH);
    float var  = ts2 * (1.f / CH) - mean * mean;
    float inv  = rsqrtf(var + 1e-6f);
#pragma unroll
    for (int i = 0; i < 5; ++i) {
        int c = tid + i * 256;
        out[(size_t)r * CH + c] = (bf16_t)((vals[i] - mean) * inv * g[c] + b[c]);
    }
}

// ---- tiled WMMA GEMM: D[M][N] = A[M][K](bf16) @ Bt[N][K](bf16)^T + bias, fused epilogues ----
// Staging via GLOBAL_LOAD_ASYNC_TO_LDS_B128 (ASYNCcnt), double-buffered.
// EPI 0: bf16 out.  1: exact GELU, bf16 out.  2: f32 out + window residual gathered from image.
// EPI 3: f32, + dense residual, scattered window->image into d_out (padding dropped).
template <int EPI>
__global__ __launch_bounds__(256) void gemm_kernel(const bf16_t* __restrict__ A,
                                                   const bf16_t* __restrict__ Bt,
                                                   const float* __restrict__ bias,
                                                   const float* __restrict__ res,
                                                   void* __restrict__ outp,
                                                   int M, int N, int K) {
    __shared__ bf16_t As[2][128][40];
    __shared__ bf16_t Bs[2][128][40];
    int tid = threadIdx.x;
    int bm = blockIdx.x, bn = blockIdx.y;
    int wave = tid >> 5, lane = tid & 31, lrow = lane & 15, lh = lane >> 4;
    int wm = wave & 1, wn = wave >> 1;

    floatx8 z = {0.f, 0.f, 0.f, 0.f, 0.f, 0.f, 0.f, 0.f};
    floatx8 acc[4][2];
#pragma unroll
    for (int mi = 0; mi < 4; ++mi)
#pragma unroll
        for (int ni = 0; ni < 2; ++ni) acc[mi][ni] = z;

    // per-thread staging slots (uniform across the wave; edge rows clamped -> deterministic)
    int rowA[2], segA[2];
    unsigned gbA[2], gbB[2];   // global byte offsets of (row, k=0) for A and B
#pragma unroll
    for (int c = 0; c < 2; ++c) {
        int idx = tid + c * 256;
        rowA[c] = idx >> 2;
        segA[c] = idx & 3;
        int gm = bm * 128 + rowA[c];
        if (gm >= M) gm = M - 1;                       // clamp: rows >= M are discarded later
        gbA[c] = (unsigned)(((size_t)gm * K + segA[c] * 8) * 2);
        int gn = bn * 128 + rowA[c];                   // N is always a multiple of 128
        gbB[c] = (unsigned)(((size_t)gn * K + segA[c] * 8) * 2);
    }

    auto issue = [&](int buf, int k0) {
        unsigned kb = (unsigned)(k0 * 2);
#pragma unroll
        for (int c = 0; c < 2; ++c) {
            unsigned la = (unsigned)(uintptr_t)&As[buf][rowA[c]][segA[c] * 8];
            async_b128(la, gbA[c] + kb, A);
            unsigned lb = (unsigned)(uintptr_t)&Bs[buf][rowA[c]][segA[c] * 8];
            async_b128(lb, gbB[c] + kb, Bt);
        }
    };

    issue(0, 0);
    wait_async0();
    __syncthreads();
    int buf = 0;
    for (int k0 = 0; k0 < K; k0 += 32) {
        bool nxt = (k0 + 32) < K;
        if (nxt) issue(buf ^ 1, k0 + 32);
        bf16x16 bfr[2];
#pragma unroll
        for (int ni = 0; ni < 2; ++ni)
            bfr[ni] = frag_b(&Bs[buf][wn * 32 + ni * 16 + lrow][0], lh * 8);
#pragma unroll
        for (int mi = 0; mi < 4; ++mi) {
            bf16x16 af = frag_a(&As[buf][wm * 64 + mi * 16 + lrow][0], lh * 8);
#pragma unroll
            for (int ni = 0; ni < 2; ++ni)
                acc[mi][ni] = wmma_bf16(af, bfr[ni], acc[mi][ni]);
        }
        if (nxt) {
            wait_async0();      // own async writes to buf^1 done
            __syncthreads();    // everyone's done (and done reading buf)
            buf ^= 1;
        }
    }

    // epilogue: element (vgpr j, lane) -> row lh*8+j, col lane&15 within a 16x16 tile
    int gn0 = bn * 128 + wn * 32 + lrow;
    float bv[2];
#pragma unroll
    for (int ni = 0; ni < 2; ++ni) bv[ni] = bias[gn0 + ni * 16];
#pragma unroll
    for (int mi = 0; mi < 4; ++mi) {
#pragma unroll
        for (int j = 0; j < 8; ++j) {
            int gm = bm * 128 + wm * 64 + mi * 16 + lh * 8 + j;
            if (gm >= M) continue;
            int ob = (EPI >= 2) ? img_base(gm) : 0;    // hoisted row -> pixel mapping
#pragma unroll
            for (int ni = 0; ni < 2; ++ni) {
                int gn = gn0 + ni * 16;
                float v = acc[mi][ni][j] + bv[ni];
                if (EPI == 0) {
                    ((bf16_t*)outp)[(size_t)gm * N + gn] = (bf16_t)v;
                } else if (EPI == 1) {
                    v = 0.5f * v * (1.f + erff(v * 0.70710678118f));
                    ((bf16_t*)outp)[(size_t)gm * N + gn] = (bf16_t)v;
                } else if (EPI == 2) {
                    float r0 = (ob >= 0) ? res[ob + gn] : 0.f;
                    ((float*)outp)[(size_t)gm * N + gn] = v + r0;
                } else {
                    float r0 = res[(size_t)gm * N + gn];
                    if (ob >= 0) ((float*)outp)[ob + gn] = v + r0;
                }
            }
        }
    }
}

// ---- fused windowed attention, one block per (window, head) ----
__global__ __launch_bounds__(256) void attn_kernel(const bf16_t* __restrict__ qkv,
                                                   const float* __restrict__ relh,
                                                   const float* __restrict__ relw,
                                                   bf16_t* __restrict__ outO) {
    extern __shared__ char smem[];
    bf16_t* qs = (bf16_t*)smem;                     // [208][104]
    bf16_t* ks = qs + A_NPAD * A_QSTR;              // [208][104]
    bf16_t* vT = ks + A_NPAD * A_QSTR;              // [96][232] (v transposed: [d][token])
    float*  S  = (float*)(vT + 96 * A_VSTR);        // [208][212] f32 scores; P bf16 aliased per-row
    _Float16* bh = (_Float16*)(S + A_NPAD * A_SSTR);// [196][14]
    _Float16* bw = bh + NTOK * WS_;                 // [196][14]

    int tid = threadIdx.x;
    int w = blockIdx.x >> 4;
    int h = blockIdx.x & 15;
    const bf16_t* base = qkv + (size_t)w * NTOK * (3 * CH);
    int cq = h * HD, ck = CH + h * HD, cv = 2 * CH + h * HD;

    // stage q, k (rows 196..207 and d 80..95 zeroed)
    for (int i = tid; i < A_NPAD * 96; i += 256) {
        int t = i / 96, d = i - t * 96;
        bf16_t qv = (bf16_t)0.f, kv = (bf16_t)0.f;
        if (t < NTOK && d < HD) {
            qv = base[t * (3 * CH) + cq + d];
            kv = base[t * (3 * CH) + ck + d];
        }
        qs[t * A_QSTR + d] = qv;
        ks[t * A_QSTR + d] = kv;
    }
    // stage v transposed (cols 196..223 and d 80..95 zeroed)
    for (int i = tid; i < 96 * A_KPAD; i += 256) {
        int d = i / A_KPAD, t = i - d * A_KPAD;
        bf16_t vv = (bf16_t)0.f;
        if (t < NTOK && d < HD) vv = base[t * (3 * CH) + cv + d];
        vT[d * A_VSTR + t] = vv;
    }
    __syncthreads();

    // decomposed relative-position bias: bh[q][ky] = q . Rh[qy-ky+13], bw[q][kx] = q . Rw[qx-kx+13]
    for (int i = tid; i < 2 * NTOK * WS_; i += 256) {
        int which = (i >= NTOK * WS_);
        int j = which ? i - NTOK * WS_ : i;
        int q = j / WS_, kc = j - q * WS_;
        int qy = q / WS_, qx = q - qy * WS_;
        const float* rv = which ? (relw + (qx - kc + WS_ - 1) * HD)
                                : (relh + (qy - kc + WS_ - 1) * HD);
        const bf16_t* qrow = qs + q * A_QSTR;
        float s = 0.f;
#pragma unroll 4
        for (int d = 0; d < HD; ++d) s += (float)qrow[d] * rv[d];
        (which ? bw : bh)[j] = (_Float16)s;
    }

    int wave = tid >> 5, lane = tid & 31, lrow = lane & 15, lh = lane >> 4;
    __syncthreads();

    // S = q @ k^T (13x13 tiles, K = 96)
    for (int tt = wave; tt < 13 * 13; tt += 8) {
        int mi = tt / 13, ni = tt - mi * 13;
        floatx8 acc = {0.f, 0.f, 0.f, 0.f, 0.f, 0.f, 0.f, 0.f};
        const bf16_t* ar = qs + (mi * 16 + lrow) * A_QSTR;
        const bf16_t* br = ks + (ni * 16 + lrow) * A_QSTR;
#pragma unroll
        for (int kk = 0; kk < 96; kk += 32) {
            bf16x16 af  = frag_a(ar, kk + lh * 8);
            bf16x16 bfv = frag_b(br, kk + lh * 8);
            acc = wmma_bf16(af, bfv, acc);
        }
        float* sp = S + (mi * 16 + lh * 8) * A_SSTR + ni * 16 + lrow;
#pragma unroll
        for (int j = 0; j < 8; ++j) sp[j * A_SSTR] = acc[j];
    }
    __syncthreads();

    // softmax with scale + bias; write P as bf16 in place (stride 2*A_SSTR bf16, no cross-row clash)
    const float sc = 0.111803398875f;  // 80^-0.5
    for (int r = wave; r < NTOK; r += 8) {
        float* srow = S + r * A_SSTR;
        float vals[7];
        float mx = -1e30f;
#pragma unroll
        for (int i = 0; i < 7; ++i) {
            int c = lane + i * 32;
            float v = -1e30f;
            if (c < NTOK)
                v = srow[c] * sc + (float)bh[r * WS_ + c / WS_] + (float)bw[r * WS_ + c % WS_];
            vals[i] = v;
            mx = fmaxf(mx, v);
        }
#pragma unroll
        for (int o = 16; o > 0; o >>= 1) mx = fmaxf(mx, __shfl_xor(mx, o, 32));
        float sum = 0.f;
#pragma unroll
        for (int i = 0; i < 7; ++i) {
            int c = lane + i * 32;
            float e = (c < NTOK) ? expf(vals[i] - mx) : 0.f;
            vals[i] = e; sum += e;
        }
#pragma unroll
        for (int o = 16; o > 0; o >>= 1) sum += __shfl_xor(sum, o, 32);
        float inv = 1.f / sum;
        bf16_t* prow = (bf16_t*)srow;
#pragma unroll
        for (int i = 0; i < 7; ++i) {
            int c = lane + i * 32;       // covers 0..223 == A_KPAD
            prow[c] = (bf16_t)(vals[i] * inv);
        }
    }
    __syncthreads();

    // O = P @ V (13 x 6 tiles, K = 224); padded P rows are exact zeros (f32 0 reinterprets to bf16 0)
    for (int tt = wave; tt < 13 * 6; tt += 8) {
        int mi = tt / 6, ni = tt - mi * 6;
        floatx8 acc = {0.f, 0.f, 0.f, 0.f, 0.f, 0.f, 0.f, 0.f};
        const bf16_t* ar = (const bf16_t*)(S + (mi * 16 + lrow) * A_SSTR);
        const bf16_t* br = vT + (ni * 16 + lrow) * A_VSTR;
#pragma unroll
        for (int kk = 0; kk < A_KPAD; kk += 32) {
            bf16x16 af  = frag_a(ar, kk + lh * 8);
            bf16x16 bfv = frag_b(br, kk + lh * 8);
            acc = wmma_bf16(af, bfv, acc);
        }
        int gd = ni * 16 + lrow;
        if (gd < HD) {
#pragma unroll
            for (int j = 0; j < 8; ++j) {
                int t = mi * 16 + lh * 8 + j;
                if (t < NTOK)
                    outO[((size_t)w * NTOK + t) * CH + h * HD + gd] = (bf16_t)acc[j];
            }
        }
    }
}

extern "C" void kernel_launch(void* const* d_in, const int* in_sizes, int n_in,
                              void* d_out, int out_size, void* d_ws, size_t ws_size,
                              hipStream_t stream) {
    (void)in_sizes; (void)n_in; (void)out_size; (void)ws_size;
    const float* img    = (const float*)d_in[0];
    const float* ln1_g  = (const float*)d_in[1];
    const float* ln1_b  = (const float*)d_in[2];
    const float* qkv_w  = (const float*)d_in[3];
    const float* qkv_b  = (const float*)d_in[4];
    const float* proj_w = (const float*)d_in[5];
    const float* proj_b = (const float*)d_in[6];
    const float* relh   = (const float*)d_in[7];
    const float* relw   = (const float*)d_in[8];
    const float* ln2_g  = (const float*)d_in[9];
    const float* ln2_b  = (const float*)d_in[10];
    const float* w1     = (const float*)d_in[11];
    const float* b1     = (const float*)d_in[12];
    const float* w2     = (const float*)d_in[13];
    const float* b2     = (const float*)d_in[14];

    char* ws = (char*)d_ws;
    size_t off = 0;
    auto arena = [&](size_t bytes) -> char* {
        char* p = ws + off;
        off += (bytes + 255) & ~(size_t)255;
        return p;
    };
    bf16_t* wt_qkv  = (bf16_t*)arena((size_t)CH * 3 * CH * 2);
    bf16_t* wt_proj = (bf16_t*)arena((size_t)CH * CH * 2);
    bf16_t* wt_w1   = (bf16_t*)arena((size_t)CH * 4 * CH * 2);
    bf16_t* wt_w2   = (bf16_t*)arena((size_t)4 * CH * CH * 2);
    bf16_t* lnbuf   = (bf16_t*)arena((size_t)NROWS * CH * 2);       // ln1 out, later ln2 out
    bf16_t* bigbuf  = (bf16_t*)arena((size_t)NROWS * 4 * CH * 2);   // qkv (3C), later mlp hidden (4C)
    bf16_t* attnO   = (bf16_t*)arena((size_t)NROWS * CH * 2);
    float*  attnres = (float*)arena((size_t)NROWS * CH * 4);        // residual-1 (attn_out)

    auto cdiv = [](int a, int b) { return (a + b - 1) / b; };

    // weights -> bf16, transposed to [N][K]
    wconv_kernel<<<cdiv(CH * 3 * CH, 256), 256, 0, stream>>>(qkv_w, wt_qkv, CH, 3 * CH);
    wconv_kernel<<<cdiv(CH * CH, 256), 256, 0, stream>>>(proj_w, wt_proj, CH, CH);
    wconv_kernel<<<cdiv(CH * 4 * CH, 256), 256, 0, stream>>>(w1, wt_w1, CH, 4 * CH);
    wconv_kernel<<<cdiv(4 * CH * CH, 256), 256, 0, stream>>>(w2, wt_w2, 4 * CH, CH);

    // LN1 (window-gather from image, padding -> 0)
    ln_kernel<<<NROWS, 256, 0, stream>>>(nullptr, img, ln1_g, ln1_b, lnbuf);

    // QKV projection (bf16 out)
    gemm_kernel<0><<<dim3(cdiv(NROWS, 128), (3 * CH) / 128), 256, 0, stream>>>(
        lnbuf, wt_qkv, qkv_b, nullptr, bigbuf, NROWS, 3 * CH, CH);

    // attention
    hipFuncSetAttribute((const void*)attn_kernel,
                        hipFuncAttributeMaxDynamicSharedMemorySize, ATTN_SMEM);
    attn_kernel<<<NWIN * NHEADS, 256, ATTN_SMEM, stream>>>(bigbuf, relh, relw, attnO);

    // output projection + residual-1 (gathered from image) -> f32 attn_out
    gemm_kernel<2><<<dim3(cdiv(NROWS, 128), CH / 128), 256, 0, stream>>>(
        attnO, wt_proj, proj_b, img, attnres, NROWS, CH, CH);

    // LN2
    ln_kernel<<<NROWS, 256, 0, stream>>>(attnres, nullptr, ln2_g, ln2_b, lnbuf);

    // MLP up + exact GELU (bf16 hidden)
    gemm_kernel<1><<<dim3(cdiv(NROWS, 128), (4 * CH) / 128), 256, 0, stream>>>(
        lnbuf, wt_w1, b1, nullptr, bigbuf, NROWS, 4 * CH, CH);

    // MLP down + residual-2, scattered window->image into d_out (padding dropped)
    gemm_kernel<3><<<dim3(cdiv(NROWS, 128), CH / 128), 256, 0, stream>>>(
        bigbuf, wt_w2, b2, attnres, d_out, NROWS, CH, 4 * CH);
}
